// MSAColumnAttention_16054587752602
// MI455X (gfx1250) — matrix operand order repaired and checked by reading, not compiled, not run on hardware
//
#include <hip/hip_runtime.h>

// MSA column attention for MI455X (gfx1250, wave32, WMMA 16x16x32 f16->f32).
// Pipeline: [prep: f32->f16 transposed weights] -> [LN + QKV/gate GEMM] ->
// [per-(r,h) attention] -> [gated output GEMM + residual].
// All shared B operands are staged once per block into the 320KB WGP LDS and
// fed to WMMA via ds_load_b128 fragments, so redundant L2 fetch is eliminated.
// Workspace requirement: (1024*256 + 256*256 + 5*256^3) f16 ~= 168.4 MB
// (fits MI455X 192MB L2, so inter-kernel traffic mostly stays on-chip).

typedef __attribute__((ext_vector_type(16))) _Float16 v16h;
typedef __attribute__((ext_vector_type(8)))  _Float16 v8h;
typedef __attribute__((ext_vector_type(8)))  float    v8f;

#define NRES 256
#define NSEQ 256
#define DNODE 256
#define NHEAD 8
#define CDIM 32

extern __shared__ unsigned char g_smem[];

// A-fragment (16x32 f16, M x K) from row-major [M][K] buffer, ld in elements.
// Lane L (m = L&15, g = L>>4): elems 0..7 -> K = k0 + g*8 + e,
//                              elems 8..15 -> K = k0 + 16 + g*8 + (e-8).
__device__ __forceinline__ v16h frag_a(const _Float16* buf, int row0, int ld, int k0) {
  const int lane = threadIdx.x & 31;
  const _Float16* p = buf + (size_t)(row0 + (lane & 15)) * ld + k0 + ((lane >> 4) * 8);
  v8h lo = *(const v8h*)(p);
  v8h hi = *(const v8h*)(p + 16);
  v16h a;
#pragma unroll
  for (int i = 0; i < 8; ++i) { a[i] = lo[i]; a[i + 8] = hi[i]; }
  return a;
}

// B-fragment (32x16 f16, K x N) read from OPERAND-TRANSPOSED row-major [N][K].
// Lane L (n = L&15, g = L>>4): elems 0..7 -> K = k0 + g*8 + e,
//                              elems 8..15 -> K = k0 + 16 + g*8 + (e-8).
__device__ __forceinline__ v16h frag_b(const _Float16* buf, int n0, int ld, int k0) {
  const int lane = threadIdx.x & 31;
  const _Float16* p = buf + (size_t)(n0 + (lane & 15)) * ld + k0 + ((lane >> 4) * 8);
  v8h lo = *(const v8h*)(p);
  v8h hi = *(const v8h*)(p + 16);
  v16h b;
#pragma unroll
  for (int i = 0; i < 8; ++i) { b[i] = lo[i]; b[i + 8] = hi[i]; }
  return b;
}

__device__ __forceinline__ v8f wmma_f16(v16h a, v16h b, v8f c) {
  return __builtin_amdgcn_wmma_f32_16x16x32_f16(false, a, false, b, (short)0, c, false, false);
}

// ---------------- kernel 0: weights f32 -> f16, operand-transposed ----------
// Wt  [1024][256]: cols 0..767 = qkv_w^T, 768..1023 = gate_w^T
// oWt [256][256] = out_w^T
__global__ void k_prep(const float* __restrict__ qkv_w, const float* __restrict__ gate_w,
                       const float* __restrict__ out_w,
                       _Float16* __restrict__ Wt, _Float16* __restrict__ oWt) {
  int idx = blockIdx.x * blockDim.x + threadIdx.x;   // (1024+256)*256 threads
  if (idx >= 1280 * 256) return;
  int n = idx >> 8, k = idx & 255;
  if (n < 768)       Wt[n * 256 + k]           = (_Float16)qkv_w[k * 768 + n];
  else if (n < 1024) Wt[n * 256 + k]           = (_Float16)gate_w[k * 256 + (n - 768)];
  else               oWt[(n - 1024) * 256 + k] = (_Float16)out_w[k * 256 + (n - 1024)];
}

// ---------------- kernel 1: LayerNorm + fused QKV/gate projection -----------
// One block per residue column r.
// LDS: Mn [256 s][256 d] f16 = 128 KB  +  staged weight tile [16 n][256 k] = 8 KB.
__global__ void k_ln_qkv(const float* __restrict__ M_raw,
                         const float* __restrict__ ln_w, const float* __restrict__ ln_b,
                         const float* __restrict__ gate_b,
                         const _Float16* __restrict__ Wt,
                         _Float16* __restrict__ qbuf, _Float16* __restrict__ kbuf,
                         _Float16* __restrict__ vT,   _Float16* __restrict__ gbuf) {
  _Float16* Mn    = (_Float16*)g_smem;                  // [256][256] f16
  _Float16* WtLds = Mn + 256 * 256;                     // [16][256]  f16
  const int r = blockIdx.x;
  const int wave = threadIdx.x >> 5, lane = threadIdx.x & 31;

  // ---- LayerNorm: one wave per row, 8 f32 per lane ----
  float w[8], b[8];
  *(float4*)(w)     = *(const float4*)(ln_w + lane * 8);
  *(float4*)(w + 4) = *(const float4*)(ln_w + lane * 8 + 4);
  *(float4*)(b)     = *(const float4*)(ln_b + lane * 8);
  *(float4*)(b + 4) = *(const float4*)(ln_b + lane * 8 + 4);
  for (int s = wave * 32; s < wave * 32 + 32; ++s) {
    const float* src = M_raw + ((size_t)s * NRES + r) * DNODE + lane * 8;
    float x[8];
    *(float4*)(x)     = *(const float4*)(src);
    *(float4*)(x + 4) = *(const float4*)(src + 4);
    float sum = 0.f, sq = 0.f;
#pragma unroll
    for (int i = 0; i < 8; ++i) { sum += x[i]; sq += x[i] * x[i]; }
#pragma unroll
    for (int m = 16; m >= 1; m >>= 1) {
      sum += __shfl_xor(sum, m, 32);
      sq  += __shfl_xor(sq,  m, 32);
    }
    float mu   = sum * (1.0f / 256.0f);
    float rstd = rsqrtf(sq * (1.0f / 256.0f) - mu * mu + 1e-5f);
    v8h o;
#pragma unroll
    for (int i = 0; i < 8; ++i) o[i] = (_Float16)((x[i] - mu) * rstd * w[i] + b[i]);
    *(v8h*)(Mn + s * 256 + lane * 8) = o;
  }

  // ---- GEMM: Mn[256 x 256] @ W[256 x 1024]; weight tile staged per tn ----
  const float qscale = 0.17677669529663687f;   // 32^-0.5
  const int n = lane & 15, hi = lane >> 4;
  const int srow = threadIdx.x >> 4;           // 0..15: weight tile row
  const int sseg = threadIdx.x & 15;           // 0..15: 16-half segment
  for (int tn = 0; tn < 64; ++tn) {
    __syncthreads();                           // WtLds reuse / Mn ready
    // stage Wt rows [tn*16, tn*16+16) x 256 K  (8 KB, 32 B per thread)
    {
      const _Float16* src = Wt + (size_t)(tn * 16 + srow) * 256 + sseg * 16;
      _Float16* dst = WtLds + srow * 256 + sseg * 16;
      *(v8h*)(dst)     = *(const v8h*)(src);
      *(v8h*)(dst + 8) = *(const v8h*)(src + 8);
    }
    __syncthreads();
    for (int tm = wave * 2; tm < wave * 2 + 2; ++tm) {
      v8f acc = {};
#pragma unroll
      for (int kt = 0; kt < 8; ++kt) {
        v16h a  = frag_a(Mn, tm * 16, 256, kt * 32);
        v16h bb = frag_b(WtLds, 0, 256, kt * 32);
        acc = wmma_f16(a, bb, acc);
      }
      const int nc = tn * 16 + n;              // global output column
#pragma unroll
      for (int j = 0; j < 8; ++j) {
        const int s = tm * 16 + hi * 8 + j;
        const float v = acc[j];
        const size_t base = ((size_t)r * 256 + s) * 256;
        if (nc < 256) {
          qbuf[base + nc] = (_Float16)(v * qscale);
        } else if (nc < 512) {
          kbuf[base + (nc - 256)] = (_Float16)v;
        } else if (nc < 768) {
          vT[(size_t)r * 65536 + (size_t)(nc - 512) * 256 + s] = (_Float16)v;  // transposed
        } else {
          float g = 1.0f / (1.0f + __expf(-(v + gate_b[nc - 768])));
          gbuf[base + (nc - 768)] = (_Float16)g;
        }
      }
    }
  }
}

// ---------------- kernel 2: attention per (head h, column r) ----------------
// K and V^T slices staged once per block in LDS; logits tiles in registers
// (16 x v8f); softmax via half-wave shuffles; weights re-striped through LDS
// for the PV GEMM. LDS: Wlds 128K + kLds 16K + vLds 16K = 160 KB.
__global__ void k_attn(const float* __restrict__ M_mask,
                       const _Float16* __restrict__ qbuf, const _Float16* __restrict__ kbuf,
                       const _Float16* __restrict__ vT,   _Float16* __restrict__ obuf) {
  _Float16* Wlds = (_Float16*)g_smem;                   // [256 q][256 k] f16
  _Float16* kLds = Wlds + 256 * 256;                    // [256 s'][32 c] f16
  _Float16* vLds = kLds + 256 * 32;                     // [32 c][256 s'] f16
  const int h = blockIdx.x, r = blockIdx.y;
  const int wave = threadIdx.x >> 5, lane = threadIdx.x & 31;
  const int n = lane & 15, hi = lane >> 4;
  const _Float16* qb = qbuf + (size_t)r * 65536;

  // stage K slice: kbuf[r][s'][h*32 .. h*32+31] -> kLds[s'][c]   (64 B/thread)
  {
    const int s = threadIdx.x;                          // 0..255
    const _Float16* src = kbuf + (size_t)r * 65536 + (size_t)s * 256 + h * 32;
#pragma unroll
    for (int i = 0; i < 4; ++i)
      *(v8h*)(kLds + s * 32 + i * 8) = *(const v8h*)(src + i * 8);
  }
  // stage V^T slice: vT[r][h*32 + c][s'] -> vLds[c][s']          (64 B/thread)
  {
    const int c = threadIdx.x >> 3, seg = threadIdx.x & 7;   // 32 rows x 8 segs
    const _Float16* src = vT + (size_t)r * 65536 + (size_t)(h * 32 + c) * 256 + seg * 32;
#pragma unroll
    for (int i = 0; i < 4; ++i)
      *(v8h*)(vLds + c * 256 + seg * 32 + i * 8) = *(const v8h*)(src + i * 8);
  }
  __syncthreads();

  for (int tm = wave * 2; tm < wave * 2 + 2; ++tm) {
    // --- logits: q(16x32) @ k^T(32x256) -> 16 tiles, K=32 => one WMMA each
    v8f accs[16];
    float bias[16];
    v16h a = frag_a(qb, tm * 16, 256, h * 32);
#pragma unroll
    for (int t = 0; t < 16; ++t) {
      bias[t] = 1.0e9f * (M_mask[(size_t)(t * 16 + n) * 256 + r] - 1.0f);
      v8f z = {};
      accs[t] = wmma_f16(a, frag_b(kLds, t * 16, 32, 0), z);
    }
    // --- softmax over k (rows live in one VGPR across a 16-lane half-group)
#pragma unroll
    for (int j = 0; j < 8; ++j) {
      float mx = -3.0e38f;
#pragma unroll
      for (int t = 0; t < 16; ++t) mx = fmaxf(mx, accs[t][j] + bias[t]);
#pragma unroll
      for (int m = 8; m >= 1; m >>= 1) mx = fmaxf(mx, __shfl_xor(mx, m, 16));
      float sum = 0.f;
#pragma unroll
      for (int t = 0; t < 16; ++t) {
        float e = __expf(accs[t][j] + bias[t] - mx);
        accs[t][j] = e;
        sum += e;
      }
#pragma unroll
      for (int m = 8; m >= 1; m >>= 1) sum += __shfl_xor(sum, m, 16);
      const float inv = 1.0f / sum;
      const int q = tm * 16 + hi * 8 + j;
#pragma unroll
      for (int t = 0; t < 16; ++t)
        Wlds[q * 256 + t * 16 + n] = (_Float16)(accs[t][j] * inv);
    }
    // --- o = weights(16x256) @ v(256x32); B from staged transposed v
#pragma unroll
    for (int tc = 0; tc < 2; ++tc) {
      v8f acc = {};
#pragma unroll
      for (int kt = 0; kt < 8; ++kt) {
        v16h aw = frag_a(Wlds, tm * 16, 256, kt * 32);
        v16h bv = frag_b(vLds, tc * 16, 256, kt * 32);
        acc = wmma_f16(aw, bv, acc);
      }
#pragma unroll
      for (int j = 0; j < 8; ++j) {
        const int q = tm * 16 + hi * 8 + j;
        obuf[((size_t)r * 256 + q) * 256 + h * 32 + tc * 16 + n] = (_Float16)acc[j];
      }
    }
  }
}

// ---------------- kernel 3: out = (o*gate) @ out_w + out_b, + residual ------
// Entire out_w^T (128 KB f16) staged once per block in LDS.
__global__ void k_out(const float* __restrict__ M_raw, const float* __restrict__ out_b,
                      const _Float16* __restrict__ oWt,
                      const _Float16* __restrict__ obuf, const _Float16* __restrict__ gbuf,
                      float* __restrict__ outp) {
  _Float16* oLds = (_Float16*)g_smem;                   // [256 d][256 hc] f16
  const int r = blockIdx.x;
  const int wave = threadIdx.x >> 5, lane = threadIdx.x & 31;
  const int n = lane & 15, hi = lane >> 4;
  const _Float16* ob = obuf + (size_t)r * 65536;
  const _Float16* gb = gbuf + (size_t)r * 65536;

  // stage out_w^T: one 256-half row per thread (512 B)
  {
    const _Float16* src = oWt + (size_t)threadIdx.x * 256;
    _Float16* dst = oLds + threadIdx.x * 256;
#pragma unroll
    for (int i = 0; i < 32; ++i)
      *(v8h*)(dst + i * 8) = *(const v8h*)(src + i * 8);
  }
  __syncthreads();

  for (int t = wave; t < 256; t += 8) {        // 16 x 16 tiles
    const int tm = t >> 4, tn = t & 15;
    v8f acc = {};
#pragma unroll
    for (int kt = 0; kt < 8; ++kt) {
      v16h ao = frag_a(ob, tm * 16, 256, kt * 32);
      v16h ag = frag_a(gb, tm * 16, 256, kt * 32);
      v16h a = ao * ag;                        // fused gating (v_pk_mul_f16)
      v16h b = frag_b(oLds, tn * 16, 256, kt * 32);
      acc = wmma_f16(a, b, acc);
    }
    const int d = tn * 16 + n;
    const float bd = out_b[d];
#pragma unroll
    for (int j = 0; j < 8; ++j) {
      const int s = tm * 16 + hi * 8 + j;
      const size_t idx = ((size_t)s * 256 + r) * 256 + d;   // output is re-swapped
      outp[idx] = M_raw[idx] + acc[j] + bd;
    }
  }
}

extern "C" void kernel_launch(void* const* d_in, const int* in_sizes, int n_in,
                              void* d_out, int out_size, void* d_ws, size_t ws_size,
                              hipStream_t stream) {
  const float* M_raw  = (const float*)d_in[0];
  const float* M_mask = (const float*)d_in[1];
  const float* ln_w   = (const float*)d_in[2];
  const float* ln_b   = (const float*)d_in[3];
  const float* qkv_w  = (const float*)d_in[4];
  const float* gate_w = (const float*)d_in[5];
  const float* gate_b = (const float*)d_in[6];
  const float* out_w  = (const float*)d_in[7];
  const float* out_b  = (const float*)d_in[8];

  // Workspace layout (f16): Wt[1024*256] | oWt[256*256] | q | k | vT | gate | o
  _Float16* Wt   = (_Float16*)d_ws;
  _Float16* oWt  = Wt + 1024 * 256;
  _Float16* qbuf = oWt + 256 * 256;
  const size_t big = (size_t)256 * 256 * 256;
  _Float16* kbuf = qbuf + big;
  _Float16* vT   = kbuf + big;
  _Float16* gbuf = vT + big;
  _Float16* obuf = gbuf + big;

  k_prep  <<<1280, 256, 0, stream>>>(qkv_w, gate_w, out_w, Wt, oWt);
  k_ln_qkv<<<256, 256, 139264, stream>>>(M_raw, ln_w, ln_b, gate_b, Wt,
                                         qbuf, kbuf, vT, gbuf);
  k_attn  <<<dim3(8, 256), 256, 163840, stream>>>(M_mask, qbuf, kbuf, vT, obuf);
  k_out   <<<256, 256, 131072, stream>>>(M_raw, out_b, oWt, obuf, gbuf, (float*)d_out);
}
